// PaiNNStageBackbone_10608569221146
// MI455X (gfx1250) — compile-verified, matrix operand order checked by reading, not run on hardware
//
#include <hip/hip_runtime.h>
#include <stdint.h>

// ---------------------------------------------------------------------------
// MI455X (gfx1250) fused PaiNN backbone:
//   node_emb = SiLU(concat(x, pos, |pos|) @ W + b)   via v_wmma_f32_16x16x32_bf16
//   graph_feat = segment_mean(node_emb, sorted batch_ids)
//
// Roofline: 33.8 GFLOP vs ~0.77 GB HBM traffic (AI ~44 FLOP/B) => needs
// ~1 PFLOP/s matmul rate at 23.3 TB/s, so bf16 WMMA (16384 FLOP/instr) with
// f32 accumulation. x staging uses gfx1250 async global->LDS DMA
// (global_load_async_to_lds_b128, ASYNCcnt) software-pipelined one tile ahead
// so global latency hides under the WMMA loop.
// ---------------------------------------------------------------------------

typedef __bf16 v16bf __attribute__((ext_vector_type(16)));
typedef float  v8f   __attribute__((ext_vector_type(8)));
typedef uint32_t u32x4 __attribute__((ext_vector_type(4)));

#define F_IN    128
#define K_REAL  132          // 128 + 3 + 1
#define KPAD    160          // padded to 5 * 32 (WMMA K)
#define KK      80           // KPAD/2 packed bf16 pairs along K
#define HID     256
#define ROWS    64           // node rows per block tile
#define WSTRIDE 260          // dword stride of sW rows (256+4: kills bank conflicts)
#define ASTRIDE 168          // ushort stride of sA rows (160+8: kills bank conflicts)
#define ADW     (ASTRIDE/2)  // sA row stride in dwords

__device__ __forceinline__ uint16_t f2bf(float f) {
  union { float f; uint32_t u; } v; v.f = f;
  uint32_t u = v.u;
  // round-to-nearest-even f32 -> bf16
  return (uint16_t)((u + 0x7FFFu + ((u >> 16) & 1u)) >> 16);
}

// Issue 64 rows x 512B of x into LDS via async DMA (8x b128 per thread).
// INST uses GVS addressing: saddr(SGPR pair) + vaddr(32b) ; vdst = LDS byte addr.
__device__ __forceinline__ void issue_x_async(const float* __restrict__ x,
                                              int base, int n_nodes,
                                              uint32_t lds_base, int tid) {
  const float* xg = x + (size_t)base * F_IN;   // uniform -> SGPR pair
  #pragma unroll
  for (int j = 0; j < 8; ++j) {
    int c = tid + j * 256;                     // 16B chunk index (0..2047)
    int row = c >> 5;                          // 32 chunks per 128-float row
    uint32_t goff = (uint32_t)c * 16u;
    if (base + row >= n_nodes) goff = 0u;      // clamp OOB rows (pad rows unused)
    uint32_t loff = lds_base + (uint32_t)c * 16u;
    asm volatile("global_load_async_to_lds_b128 %0, %1, %2"
                 :: "v"(loff), "v"(goff), "s"(xg) : "memory");
  }
}

__global__ __launch_bounds__(256) void painn_fused(
    const float* __restrict__ x, const float* __restrict__ pos,
    const float* __restrict__ W, const float* __restrict__ bias,
    float* __restrict__ out, int n_nodes, int n_tiles)
{
  // W staged as bf16 K-pairs: sW[kk][n] = bf16(W[2kk][n]) | bf16(W[2kk+1][n])<<16
  __shared__ uint32_t sW[KK * WSTRIDE];          // 83,200 B
  __shared__ uint16_t sA[ROWS * ASTRIDE];        // 21,504 B (bf16 feats tile)
  __shared__ float    xstage[ROWS * F_IN];       // 32,768 B (async f32 landing pad)

  const int tid  = threadIdx.x;
  const int lane = tid & 31;
  const int wid  = tid >> 5;          // 8 waves
  const int mhalf = wid >> 2;         // wave owns rows [32*mhalf, 32*mhalf+32)
  const int nq    = wid & 3;          // wave owns cols [64*nq, 64*nq+64)
  const int l16   = lane & 15;
  const bool lo_half = (lane < 16);
  const uint32_t xstage_off = (uint32_t)(uintptr_t)&xstage[0];   // LDS byte addr

  // ---- stage W once per block (f32 -> packed bf16 pairs, zero-pad K) ----
  for (int idx = tid; idx < KK * HID; idx += 256) {
    int kk = idx >> 8;
    int n  = idx & 255;
    int k0 = 2 * kk, k1 = 2 * kk + 1;
    float w0 = (k0 < K_REAL) ? W[k0 * HID + n] : 0.0f;
    float w1 = (k1 < K_REAL) ? W[k1 * HID + n] : 0.0f;
    sW[kk * WSTRIDE + n] = (uint32_t)f2bf(w0) | ((uint32_t)f2bf(w1) << 16);
  }
  // ---- zero sA K-pad columns (132..167) once; never rewritten ----
  for (int idx = tid; idx < ROWS * (ASTRIDE - K_REAL); idx += 256) {
    int r = idx / (ASTRIDE - K_REAL);
    int c = K_REAL + (idx - r * (ASTRIDE - K_REAL));
    sA[r * ASTRIDE + c] = 0;
  }

  // ---- prologue: async-stage first tile's x ----
  if (blockIdx.x < (unsigned)n_tiles)
    issue_x_async(x, blockIdx.x * ROWS, n_nodes, xstage_off, tid);

  for (int tile = blockIdx.x; tile < n_tiles; tile += gridDim.x) {
    const int base = tile * ROWS;

    asm volatile("s_wait_asynccnt 0x0" ::: "memory");  // xstage holds this tile
    __syncthreads();                                   // + sA free of prev readers

    // ---- pack xstage f32 -> sA bf16 pairs (2 elements per thread-iter) ----
    for (int i = tid; i < ROWS * (F_IN / 2); i += 256) {
      int r  = i >> 6;
      int cp = i & 63;
      float2 v = *reinterpret_cast<const float2*>(&xstage[r * F_IN + 2 * cp]);
      uint32_t pk = (uint32_t)f2bf(v.x) | ((uint32_t)f2bf(v.y) << 16);
      reinterpret_cast<uint32_t*>(sA)[r * ADW + cp] = pk;
    }
    // ---- pos + |pos| for this tile (cols 128..131) ----
    if (tid < ROWS) {
      int node = base + tid;
      float px = 0.f, py = 0.f, pz = 0.f, rad = 0.f;
      if (node < n_nodes) {
        px = pos[(size_t)node * 3 + 0];
        py = pos[(size_t)node * 3 + 1];
        pz = pos[(size_t)node * 3 + 2];
        rad = sqrtf(px * px + py * py + pz * pz);
      }
      uint32_t* d = reinterpret_cast<uint32_t*>(sA) + tid * ADW + (F_IN / 2);
      d[0] = (uint32_t)f2bf(px) | ((uint32_t)f2bf(py) << 16);
      d[1] = (uint32_t)f2bf(pz) | ((uint32_t)f2bf(rad) << 16);
    }
    __syncthreads();

    // ---- overlap: DMA next tile's x while we run the WMMA loop ----
    {
      int nxt = tile + gridDim.x;
      if (nxt < n_tiles) issue_x_async(x, nxt * ROWS, n_nodes, xstage_off, tid);
    }

    v8f acc[2][4];
    #pragma unroll
    for (int m = 0; m < 2; ++m)
      #pragma unroll
      for (int t = 0; t < 4; ++t)
        acc[m][t] = (v8f){0.f,0.f,0.f,0.f,0.f,0.f,0.f,0.f};

    const int kb_lo = lo_half ? 0 : 8;     // 16-bit A layout half-wave K split

    #pragma unroll
    for (int ks = 0; ks < KPAD / 32; ++ks) {
      // ---- A fragments (2 M-tiles), each = 2x ds_load_b128 from sA ----
      v16bf afrag[2];
      #pragma unroll
      for (int m = 0; m < 2; ++m) {
        int row = 32 * mhalf + 16 * m + l16;
        const uint16_t* p = &sA[row * ASTRIDE + ks * 32 + kb_lo];
        struct Q { u32x4 a, b; } q;
        q.a = *reinterpret_cast<const u32x4*>(p);        // K kb..kb+7
        q.b = *reinterpret_cast<const u32x4*>(p + 16);   // K kb+16..kb+23
        afrag[m] = __builtin_bit_cast(v16bf, q);
      }
      // ---- 4 N-tiles: load B fragment once, reuse across both M-tiles ----
      #pragma unroll
      for (int t = 0; t < 4; ++t) {
        int col = 64 * nq + 16 * t + l16;
        int kk0 = ks * 16 + (lo_half ? 0 : 4);
        struct B8 { uint32_t u[8]; } bq;
        #pragma unroll
        for (int i = 0; i < 4; ++i) {
          bq.u[i]     = sW[(kk0 + i)     * WSTRIDE + col];
          bq.u[4 + i] = sW[(kk0 + 8 + i) * WSTRIDE + col];
        }
        v16bf bfrag = __builtin_bit_cast(v16bf, bq);
        acc[0][t] = __builtin_amdgcn_wmma_f32_16x16x32_bf16(
            false, afrag[0], false, bfrag, (short)0, acc[0][t], false, false);
        acc[1][t] = __builtin_amdgcn_wmma_f32_16x16x32_bf16(
            false, afrag[1], false, bfrag, (short)0, acc[1][t], false, false);
      }
    }

    // ---- epilogue: +bias, SiLU, coalesced stores ----
    #pragma unroll
    for (int t = 0; t < 4; ++t) {
      int col = 64 * nq + 16 * t + l16;
      float bv = bias[col];
      #pragma unroll
      for (int m = 0; m < 2; ++m) {
        #pragma unroll
        for (int r = 0; r < 8; ++r) {
          int row = base + 32 * mhalf + 16 * m + r + (lo_half ? 0 : 8);
          if (row < n_nodes) {
            float v = acc[m][t][r] + bv;
            float s = v / (1.0f + __expf(-v));     // SiLU
            out[(size_t)row * HID + col] = s;
          }
        }
      }
    }
  }
}

// ---- deterministic segment-mean pooling over sorted batch_ids ----
__global__ __launch_bounds__(256) void painn_pool(
    const float* __restrict__ emb, const int* __restrict__ bids,
    float* __restrict__ gout, int n_nodes)
{
  __shared__ int s_lo, s_hi;
  const int g = blockIdx.x;
  if (threadIdx.x < 2) {
    int target = g + (int)threadIdx.x;    // lower_bound(bids, target)
    int lo = 0, hi = n_nodes;
    while (lo < hi) {
      int mid = (lo + hi) >> 1;
      if (bids[mid] < target) lo = mid + 1; else hi = mid;
    }
    if (threadIdx.x == 0) s_lo = lo; else s_hi = lo;
  }
  __syncthreads();
  const int lo = s_lo, hi = s_hi;
  float sum = 0.0f;
  for (int i = lo; i < hi; ++i)
    sum += emb[(size_t)i * HID + threadIdx.x];
  float cnt = (float)(hi - lo);
  if (cnt < 1.0f) cnt = 1.0f;              // empty-graph guard (matches reference)
  gout[(size_t)g * HID + threadIdx.x] = sum / cnt;
}

extern "C" void kernel_launch(void* const* d_in, const int* in_sizes, int n_in,
                              void* d_out, int out_size, void* d_ws, size_t ws_size,
                              hipStream_t stream) {
  const float* x    = (const float*)d_in[0];
  const float* pos  = (const float*)d_in[1];
  const int*   bids = (const int*)d_in[2];
  const float* W    = (const float*)d_in[3];
  const float* b    = (const float*)d_in[4];

  const int n_nodes = in_sizes[2];                       // batch_ids length
  const int num_graphs = (out_size - n_nodes * HID) / HID;

  float* out_node  = (float*)d_out;
  float* out_graph = out_node + (size_t)n_nodes * HID;

  const int n_tiles = (n_nodes + ROWS - 1) / ROWS;
  int nb = n_tiles < 1024 ? n_tiles : 1024;              // persistent blocks

  painn_fused<<<nb, 256, 0, stream>>>(x, pos, W, b, out_node, n_nodes, n_tiles);
  painn_pool<<<num_graphs, 256, 0, stream>>>(out_node, bids, out_graph, n_nodes);
}
